// Model_38113539785078
// MI455X (gfx1250) — matrix operand order repaired and checked by reading, not compiled, not run on hardware
//
#include <hip/hip_runtime.h>
#include <hip/hip_bf16.h>

typedef __attribute__((ext_vector_type(16))) _Float16 v16h;
typedef __attribute__((ext_vector_type(8)))  _Float16 v8h;
typedef __attribute__((ext_vector_type(8)))  float    v8f;

__device__ inline v8f wmma_f16(v16h a, v16h b, v8f c) {
  // D = A(16x32 f16) x B(32x16 f16) + C(16x16 f32)
  return __builtin_amdgcn_wmma_f32_16x16x32_f16(
      /*neg_a=*/false, a, /*neg_b=*/false, b,
      /*c_mod=*/(short)0, c, /*reuse_a=*/false, /*reuse_b=*/false);
}

// ---------------------------------------------------------------------------
// MHSA + mean over L=8 words, d=64, H=4 heads (hd=16).
// 2 waves/block, 2 docs/wave -> 4 docs/block. Operands staged in LDS in WMMA
// fragment-register order so every v16h fragment is two ds_load_b128.
// ---------------------------------------------------------------------------
#define MHSA_WAVES 2
#define MHSA_TPB   (MHSA_WAVES * 32)

__global__ __launch_bounds__(MHSA_TPB)
void mhsa_mean_kernel(const float* __restrict__ W_word,
                      const int*   __restrict__ ids,      // [Ndoc, 8]
                      const float* __restrict__ Wq, const float* __restrict__ Wk,
                      const float* __restrict__ Wv, const float* __restrict__ Wo,
                      float* __restrict__ out,            // [Ndoc, 64]
                      int Ndoc)
{
  // B fragments: sWB[w][kstep][tn][lane][i] ; lane slot = 16 halfs = 32 B
  __shared__ __align__(16) _Float16 sWB[4][2][4][32][16];        // 32 KB
  // A fragments: sXA[wave][kstep][lane][i]
  __shared__ __align__(16) _Float16 sXA[MHSA_WAVES][2][32][16];  // 4 KB
  __shared__ float sQ[MHSA_WAVES][16][68];                       // 8.5 KB each
  __shared__ float sK[MHSA_WAVES][16][68];
  __shared__ float sV[MHSA_WAVES][16][68];

  const int tid  = threadIdx.x;
  const int wave = tid >> 5;
  const int lane = tid & 31;

  // ---- stage weights pre-swizzled into B-fragment layout (once per block) ----
  // B layout (32x16 f16 per fragment): lane = n + 16*((K>>3)&1), i = (K&7) + 8*((K>>4)&1).
  for (int f = tid; f < 4 * 64 * 64; f += MHSA_TPB) {
    const int w = f >> 12, j = f & 4095;
    const int K = j >> 6, N = j & 63;            // W is [K=64][N=64] row-major
    const float v = (w == 0 ? Wq : w == 1 ? Wk : w == 2 ? Wv : Wo)[j];
    const int ln = (N & 15) + (((K >> 3) & 1) << 4);
    sWB[w][K >> 5][N >> 4][ln][(K & 7) + (((K >> 4) & 1) << 3)] = (_Float16)v;
  }

  const int docBase = blockIdx.x * (MHSA_WAVES * 2) + wave * 2;

  // ---- gather X (16 rows = 2 docs x 8 words), write in A-fragment layout ----
  // A layout (16x32 f16): slot lane = m + 16*g with g=(krel>>3)&1,
  // i = (krel&7) + 8*((krel>>4)&1)  => any aligned 8-col run is contiguous.
  {
    const int row = lane & 15;
    const int c0  = (lane < 16) ? 0 : 32;        // this lane covers cols c0..c0+31
    const int doc = min(docBase + (row >> 3), Ndoc - 1);
    const int wid = ids[doc * 8 + (row & 7)];
    const float4* src = reinterpret_cast<const float4*>(W_word + (size_t)wid * 64 + c0);
    _Float16 h[32];
#pragma unroll
    for (int v = 0; v < 8; ++v) {
      float4 fv = src[v];
      h[v * 4 + 0] = (_Float16)fv.x;
      h[v * 4 + 1] = (_Float16)fv.y;
      h[v * 4 + 2] = (_Float16)fv.z;
      h[v * 4 + 3] = (_Float16)fv.w;
    }
    const int ks = c0 >> 5;
#pragma unroll
    for (int r = 0; r < 4; ++r) {
      const int c  = c0 + r * 8;
      const int g  = (c >> 3) & 1;
      const int ib = ((c >> 4) & 1) * 8;
      v8h chunk;
#pragma unroll
      for (int j = 0; j < 8; ++j) chunk[j] = h[r * 8 + j];
      *reinterpret_cast<v8h*>(&sXA[wave][ks][row + g * 16][ib]) = chunk;
    }
  }
  __syncthreads();

  auto loadA = [&](int ks) -> v16h {
    return *reinterpret_cast<const v16h*>(&sXA[wave][ks][lane][0]);
  };
  auto loadB = [&](int w, int ks, int tn) -> v16h {
    return *reinterpret_cast<const v16h*>(&sWB[w][ks][tn][lane][0]);
  };

  const v16h a0 = loadA(0);
  const v16h a1 = loadA(1);

  // ---- Q/K/V projections via WMMA; spill D-layout to LDS f32 ----
#pragma unroll
  for (int w = 0; w < 3; ++w) {
    float* dst = (w == 0) ? &sQ[wave][0][0] : (w == 1) ? &sK[wave][0][0] : &sV[wave][0][0];
#pragma unroll
    for (int tn = 0; tn < 4; ++tn) {
      v8f c = {};
      c = wmma_f16(a0, loadB(w, 0, tn), c);
      c = wmma_f16(a1, loadB(w, 1, tn), c);
      const int mbase = (lane < 16) ? 0 : 8;     // D layout: VGPR r -> M=mbase+r
      const int n     = tn * 16 + (lane & 15);
#pragma unroll
      for (int r = 0; r < 8; ++r)
        dst[(mbase + r) * 68 + n] = c[r];
    }
  }
  __syncthreads();

  // ---- per-(doc,head,row) attention in VALU: 64 rows of work, 2 per lane ----
  float o[2][16];
#pragma unroll
  for (int p = 0; p < 2; ++p) {
    const int idx  = lane + 32 * p;     // 0..63
    const int doc  = idx >> 5;          // local doc 0/1
    const int rem  = idx & 31;
    const int head = rem >> 3;
    const int l    = rem & 7;
    const int row  = doc * 8 + l;
    const float* q = &sQ[wave][row][head * 16];
    float s[8], mx = -1e30f;
#pragma unroll
    for (int m = 0; m < 8; ++m) {
      const float* kr = &sK[wave][doc * 8 + m][head * 16];
      float acc = 0.f;
#pragma unroll
      for (int j = 0; j < 16; ++j) acc += q[j] * kr[j];
      s[m] = acc * 0.25f;               // 1/sqrt(hd), hd=16
      mx = fmaxf(mx, s[m]);
    }
    float den = 0.f;
#pragma unroll
    for (int m = 0; m < 8; ++m) { s[m] = __expf(s[m] - mx); den += s[m]; }
    const float inv = 1.0f / den;
#pragma unroll
    for (int j = 0; j < 16; ++j) {
      float acc = 0.f;
#pragma unroll
      for (int m = 0; m < 8; ++m) acc += s[m] * sV[wave][doc * 8 + m][head * 16 + j];
      o[p][j] = acc * inv;
    }
  }
  __syncthreads();
  // write attention output O straight into A-fragment layout (2 b128 stores / p)
#pragma unroll
  for (int p = 0; p < 2; ++p) {
    const int idx  = lane + 32 * p;
    const int row  = (idx >> 5) * 8 + (idx & 7);
    const int head = (idx & 31) >> 3;
    const int ks   = head >> 1;
    const int ib   = (head & 1) * 8;
    v8h lo, hi;
#pragma unroll
    for (int j = 0; j < 8; ++j) {
      lo[j] = (_Float16)o[p][j];
      hi[j] = (_Float16)o[p][8 + j];
    }
    *reinterpret_cast<v8h*>(&sXA[wave][ks][row][ib])      = lo;  // cols j=0..7  (g=0)
    *reinterpret_cast<v8h*>(&sXA[wave][ks][row + 16][ib]) = hi;  // cols j=8..15 (g=1)
  }
  __syncthreads();

  // ---- O @ Wo via WMMA; mean over L=8 collapses in D layout ----
  const v16h oa0 = loadA(0);
  const v16h oa1 = loadA(1);
#pragma unroll
  for (int tn = 0; tn < 4; ++tn) {
    v8f c = {};
    c = wmma_f16(oa0, loadB(3, 0, tn), c);
    c = wmma_f16(oa1, loadB(3, 1, tn), c);
    float sum = 0.f;
#pragma unroll
    for (int r = 0; r < 8; ++r) sum += c[r];   // rows 0..7 (doc0) or 8..15 (doc1)
    const int d = docBase + ((lane < 16) ? 0 : 1);
    if (d < Ndoc)
      out[(size_t)d * 64 + tn * 16 + (lane & 15)] = sum * 0.125f;
  }
}

// ---------------------------------------------------------------------------
// Graph / segment-sum kernels (bandwidth + atomic bound; no matrix structure)
// ---------------------------------------------------------------------------
__global__ void degrees_kernel(const int* __restrict__ profile_dst, int R,
                               const int* __restrict__ p_src,
                               const int* __restrict__ p_dst, int E,
                               float* __restrict__ deg_p, float* __restrict__ deg_i)
{
  const int t = blockIdx.x * blockDim.x + threadIdx.x;
  if (t < R) atomicAdd(&deg_p[profile_dst[t]], 1.0f);
  if (t < E) {
    atomicAdd(&deg_i[p_src[t]], 1.0f);
    atomicAdd(&deg_i[p_dst[t]], 1.0f);
  }
}

__global__ void scatter_reviews(const float* __restrict__ h_review,
                                const int* __restrict__ dst, int R,
                                float* __restrict__ ent_h)
{
  const long long t = (long long)blockIdx.x * blockDim.x + threadIdx.x;
  const long long row = t >> 6;
  const int col = (int)(t & 63);
  if (row < R)
    atomicAdd(&ent_h[(size_t)dst[row] * 64 + col], h_review[(size_t)row * 64 + col]);
}

__global__ void build_e0(const float* __restrict__ W_entity,
                         const float* __restrict__ ent_h,
                         const float* __restrict__ deg_p,
                         const float* __restrict__ deg_i,
                         float* __restrict__ e0, float* __restrict__ inv_s, int N)
{
  const long long t = (long long)blockIdx.x * blockDim.x + threadIdx.x;
  const long long n = t >> 7;
  const int c = (int)(t & 127);
  if (n >= N) return;
  const float v = (c < 64) ? W_entity[n * 64 + c]
                           : ent_h[n * 64 + (c - 64)] / fmaxf(deg_p[n], 1.0f);
  e0[n * 128 + c] = v;
  if (c == 0) inv_s[n] = rsqrtf(fmaxf(deg_i[n], 1.0f));
}

__global__ void build_qe0(const float* __restrict__ W_query,
                          const float* __restrict__ h_query,
                          float* __restrict__ q_e0, int Q)
{
  const long long t = (long long)blockIdx.x * blockDim.x + threadIdx.x;
  const long long n = t >> 7;
  const int c = (int)(t & 127);
  if (n >= Q) return;
  q_e0[n * 128 + c] = (c < 64) ? W_query[n * 64 + c] : h_query[n * 64 + (c - 64)];
}

__global__ void conv_kernel(const float* __restrict__ e0,
                            const float* __restrict__ q_e0,
                            const int* __restrict__ p_src,
                            const int* __restrict__ p_dst,
                            const int* __restrict__ p_qid,
                            const float* __restrict__ inv_s,
                            float* __restrict__ agg, int E)
{
  const long long t = (long long)blockIdx.x * blockDim.x + threadIdx.x;
  const long long e = t >> 7;
  const int c = (int)(t & 127);
  if (e >= E) return;
  const int s = p_src[e], d = p_dst[e], q = p_qid[e];
  // purchased: e_k[src] + q_e0[qid] * inv_sqrt[src] -> dst
  const float m = e0[(size_t)s * 128 + c] + q_e0[(size_t)q * 128 + c] * inv_s[s];
  atomicAdd(&agg[(size_t)d * 128 + c], m);
  // purchased_by: e_k[dst] * inv_sqrt[dst] -> src
  atomicAdd(&agg[(size_t)s * 128 + c], e0[(size_t)d * 128 + c] * inv_s[d]);
}

__global__ void finalize_e(const float* __restrict__ e0,
                           const float* __restrict__ agg,
                           const float* __restrict__ inv_s,
                           float* __restrict__ e, int N)
{
  const long long t = (long long)blockIdx.x * blockDim.x + threadIdx.x;
  const long long n = t >> 7;
  const int c = (int)(t & 127);
  if (n >= N) return;
  e[n * 128 + c] = 0.5f * (e0[n * 128 + c] + agg[n * 128 + c] * inv_s[n]);
}

__global__ void batch_out(const float* __restrict__ e,
                          const float* __restrict__ qb,
                          const int* __restrict__ users,
                          const int* __restrict__ items,
                          const int* __restrict__ negs,
                          float* __restrict__ out, int B)
{
  const long long t = (long long)blockIdx.x * blockDim.x + threadIdx.x;
  const long long n = t >> 7;
  const int c = (int)(t & 127);
  if (n >= B) return;
  const float pq = (c < 64) ? 0.0f : qb[n * 64 + (c - 64)];
  out[(size_t)n * 128 + c]           = e[(size_t)users[n] * 128 + c] + pq;
  out[(size_t)(B + n) * 128 + c]     = e[(size_t)items[n] * 128 + c];
  out[(size_t)(2 * B + n) * 128 + c] = e[(size_t)negs[n] * 128 + c];
}

// ---------------------------------------------------------------------------
extern "C" void kernel_launch(void* const* d_in, const int* in_sizes, int n_in,
                              void* d_out, int out_size, void* d_ws, size_t ws_size,
                              hipStream_t stream)
{
  const float* W_word   = (const float*)d_in[0];
  const float* W_query  = (const float*)d_in[1];
  const float* W_entity = (const float*)d_in[2];
  const float* Wq = (const float*)d_in[3];
  const float* Wk = (const float*)d_in[4];
  const float* Wv = (const float*)d_in[5];
  const float* Wo = (const float*)d_in[6];
  const int* query_word_ids  = (const int*)d_in[7];
  const int* review_word_ids = (const int*)d_in[8];
  const int* profile_dst = (const int*)d_in[9];
  const int* p_src = (const int*)d_in[10];
  const int* p_dst = (const int*)d_in[11];
  const int* p_qid = (const int*)d_in[12];
  const int* users = (const int*)d_in[13];
  const int* items = (const int*)d_in[14];
  const int* negs  = (const int*)d_in[15];
  const int* query_words = (const int*)d_in[16];

  const int Q = in_sizes[1] / 64;   // 50000
  const int N = in_sizes[2] / 64;   // 100000
  const int R = in_sizes[8] / 8;    // 100000
  const int E = in_sizes[10];       // 2000000
  const int B = in_sizes[13];       // 1024
  float* out = (float*)d_out;
  (void)n_in; (void)out_size; (void)ws_size;

  // workspace carve-out
  char* ws = (char*)d_ws;
  size_t off = 0;
  auto carve = [&](size_t bytes) -> float* {
    float* p = (float*)(ws + off);
    off += (bytes + 255) & ~(size_t)255;
    return p;
  };
  float* h_query = carve((size_t)Q * 64 * 4);
  float* h_review= carve((size_t)R * 64 * 4);
  float* qb      = carve((size_t)B * 64 * 4);
  float* deg_p   = carve((size_t)N * 4);
  float* deg_i   = carve((size_t)N * 4);
  float* ent_h   = carve((size_t)N * 64 * 4);
  float* e0      = carve((size_t)N * 128 * 4);
  float* q_e0    = carve((size_t)Q * 128 * 4);
  float* agg     = carve((size_t)N * 128 * 4);
  float* inv_s   = carve((size_t)N * 4);
  float* e_fin   = carve((size_t)N * 128 * 4);

  // zero accumulation buffers each call (deterministic across graph replays)
  hipMemsetAsync(deg_p, 0, (size_t)N * 4, stream);
  hipMemsetAsync(deg_i, 0, (size_t)N * 4, stream);
  hipMemsetAsync(ent_h, 0, (size_t)N * 64 * 4, stream);
  hipMemsetAsync(agg,   0, (size_t)N * 128 * 4, stream);

  // --- WMMA MHSA+mean: 4 docs per 64-thread block ---
  mhsa_mean_kernel<<<(Q + 3) / 4, MHSA_TPB, 0, stream>>>(W_word, query_word_ids,  Wq, Wk, Wv, Wo, h_query, Q);
  mhsa_mean_kernel<<<(R + 3) / 4, MHSA_TPB, 0, stream>>>(W_word, review_word_ids, Wq, Wk, Wv, Wo, h_review, R);
  mhsa_mean_kernel<<<(B + 3) / 4, MHSA_TPB, 0, stream>>>(W_word, query_words,     Wq, Wk, Wv, Wo, qb, B);

  const int T = (R > E) ? R : E;
  degrees_kernel<<<(T + 255) / 256, 256, 0, stream>>>(profile_dst, R, p_src, p_dst, E, deg_p, deg_i);
  scatter_reviews<<<(unsigned)(((size_t)R * 64 + 255) / 256), 256, 0, stream>>>(h_review, profile_dst, R, ent_h);
  build_e0<<<(unsigned)(((size_t)N * 128 + 255) / 256), 256, 0, stream>>>(W_entity, ent_h, deg_p, deg_i, e0, inv_s, N);
  build_qe0<<<(unsigned)(((size_t)Q * 128 + 255) / 256), 256, 0, stream>>>(W_query, h_query, q_e0, Q);
  conv_kernel<<<(unsigned)(((size_t)E * 128 + 255) / 256), 256, 0, stream>>>(e0, q_e0, p_src, p_dst, p_qid, inv_s, agg, E);
  finalize_e<<<(unsigned)(((size_t)N * 128 + 255) / 256), 256, 0, stream>>>(e0, agg, inv_s, e_fin, N);
  batch_out<<<(unsigned)(((size_t)B * 128 + 255) / 256), 256, 0, stream>>>(e_fin, qb, users, items, negs, out, B);
}
